// Siamese_65919158059018
// MI455X (gfx1250) — matrix operand order repaired and checked by reading, not compiled, not run on hardware
//
#include <hip/hip_runtime.h>
#include <hip/hip_bf16.h>

// ---------------------------------------------------------------------------
// Types for CDNA5 WMMA (wave32) + TDM descriptors
// ---------------------------------------------------------------------------
typedef __attribute__((ext_vector_type(16))) __bf16 bf16x16;
typedef __attribute__((ext_vector_type(8)))  float  f32x8;
typedef unsigned int u32x4 __attribute__((ext_vector_type(4)));
typedef int          i32x4 __attribute__((ext_vector_type(4)));
typedef int          i32x8 __attribute__((ext_vector_type(8)));

__device__ __forceinline__ f32x8 wmma_bf16(bf16x16 a, bf16x16 b, f32x8 c) {
  return __builtin_amdgcn_wmma_f32_16x16x32_bf16(false, a, false, b,
                                                 (short)0, c, false, false);
}

__device__ __forceinline__ float sigmoidf(float x) {
  return 1.0f / (1.0f + __expf(-x));
}

__device__ __forceinline__ unsigned int rfl(unsigned int x) {
  return (unsigned int)__builtin_amdgcn_readfirstlane((int)x);
}

// ---------------------------------------------------------------------------
// TDM: DMA a 2D bf16 tile [rows x rowlen] (row stride = stride_elems) from
// global into LDS, with 1-DWORD pad after every 16 DWORDs (64B) so the LDS
// image has a 68B row stride == __bf16[rows][34]. ISA cdna5 ch.8 D# layout.
// ---------------------------------------------------------------------------
__device__ __forceinline__ void tdm_load_tile_bf16(
    unsigned int lds_byte_addr, const void* gptr,
    unsigned int rows, unsigned int rowlen, unsigned long long stride_elems) {
  unsigned long long ga = (unsigned long long)gptr;
  // group 0: count=1 | lds_addr | global_addr[56:0] | type=2
  unsigned int g0w0 = rfl(1u);
  unsigned int g0w1 = rfl(lds_byte_addr);
  unsigned int g0w2 = rfl((unsigned int)ga);
  unsigned int g0w3 = rfl(((unsigned int)(ga >> 32) & 0x1FFFFFFu) | (2u << 30));
  // group 1: data_size=1 (2B), pad_enable, pad_interval=3 (16 DW), pad_amount=0 (1 DW)
  unsigned int w0 = rfl((1u << 16) | (1u << 20) | (3u << 22));
  unsigned int td0 = rowlen, td1 = rows;
  unsigned int w1 = rfl((td0 & 0xFFFFu) << 16);                 // tensor_dim0 lo
  unsigned int w2 = rfl((td0 >> 16) | ((td1 & 0xFFFFu) << 16)); // td0 hi | td1 lo
  unsigned int w3 = rfl((td1 >> 16) | (rowlen << 16));          // td1 hi | tile_dim0
  unsigned int w4 = rfl(rows);                                  // tile_dim1 (tile_dim2=0)
  unsigned int w5 = rfl((unsigned int)stride_elems);            // dim0_stride lo
  unsigned int w6 = rfl((unsigned int)(stride_elems >> 32) & 0xFFFFu);
  unsigned int w7 = rfl(0u);

  u32x4 g0 = {g0w0, g0w1, g0w2, g0w3};
  i32x8 g1 = {(int)w0, (int)w1, (int)w2, (int)w3,
              (int)w4, (int)w5, (int)w6, (int)w7};
  i32x4 z4 = {0, 0, 0, 0};
  i32x8 z8 = {0, 0, 0, 0, 0, 0, 0, 0};
  // clang-23 / therock arity: (g0, g1, g2, g3, extra, cpol)
  __builtin_amdgcn_tensor_load_to_lds(g0, g1, z4, z4, z8, 0);
}

// ---------------------------------------------------------------------------
// Problem constants
// ---------------------------------------------------------------------------
#define TT      128            // sequence length
#define EE      768            // embedding
#define H1      128            // lstm1 hidden
#define H2      64             // lstm2 hidden
#define NSEQP   928            // 920 sequences padded to multiple of 16
#define MPAD    (NSEQP * TT)   // 118784 padded token rows (multiple of 64)

// ---------------------------------------------------------------------------
// Kernel: fp32 [K][N] weights -> bf16 transposed [N][K]
// ---------------------------------------------------------------------------
__global__ void k_transpose_bf16(const float* __restrict__ in,
                                 __bf16* __restrict__ out, int K, int N) {
  int idx = blockIdx.x * blockDim.x + threadIdx.x;
  if (idx >= K * N) return;
  int k = idx / N, n = idx % N;
  out[(long)n * K + k] = (__bf16)in[(long)k * N + n];
}

// ---------------------------------------------------------------------------
// Kernel: gather the 4 input chunks into one padded bf16 matrix [MPAD][EE]
// (pad rows clamp into model_ans -> deterministic)
// ---------------------------------------------------------------------------
__global__ void k_gather_bf16(const float* __restrict__ A0,
                              const float* __restrict__ A1,
                              const float* __restrict__ A2,
                              const float* __restrict__ A3,
                              __bf16* __restrict__ out) {
  long idx = (long)blockIdx.x * blockDim.x + threadIdx.x;
  if (idx >= (long)MPAD * EE) return;
  long row = idx / EE;
  int  col = (int)(idx % EE);
  const float* base; long local;
  if (row < (long)200 * TT)      { base = A0; local = row; }
  else if (row < (long)400 * TT) { base = A1; local = row - (long)200 * TT; }
  else if (row < (long)912 * TT) { base = A2; local = row - (long)400 * TT; }
  else { base = A3; local = row - (long)912 * TT;
         if (local >= (long)8 * TT) local = (long)8 * TT - 1; }
  out[idx] = (__bf16)base[local * EE + col];
}

// ---------------------------------------------------------------------------
// Kernel: C[M][N] = A[M][K] @ B + bias; A bf16 [M][K], B bf16 transposed [N][K].
// 64x64 tile per 4-wave workgroup. Double-buffered TDM feeds LDS
// (tensor_load_to_lds + s_wait_tensorcnt), WMMA consumes.
// ---------------------------------------------------------------------------
__global__ void __launch_bounds__(128)
k_gemm_tdm(const __bf16* __restrict__ A, const __bf16* __restrict__ BT,
           const float* __restrict__ bias, float* __restrict__ C,
           int K, int N) {
  __shared__ __bf16 As[2][64][34];   // TDM pad => 68B row stride
  __shared__ __bf16 Bs[2][64][34];

  const int tid  = threadIdx.x;
  const int lane = tid & 31;
  const int wv   = tid >> 5;
  const long r0  = (long)blockIdx.y * 64;
  const int  c0  = blockIdx.x * 64;

  f32x8 acc[4];
#pragma unroll
  for (int j = 0; j < 4; ++j)
    acc[j] = (f32x8){0.f, 0.f, 0.f, 0.f, 0.f, 0.f, 0.f, 0.f};

  const int nsteps = K / 32;

  auto issue = [&](int s, int buf) {
    tdm_load_tile_bf16((unsigned int)(uintptr_t)&As[buf][0][0],
                       A + r0 * K + s * 32, 64, 32, (unsigned long long)K);
    tdm_load_tile_bf16((unsigned int)(uintptr_t)&Bs[buf][0][0],
                       BT + (long)c0 * K + s * 32, 64, 32, (unsigned long long)K);
  };

  if (wv == 0) issue(0, 0);

  for (int s = 0; s < nsteps; ++s) {
    const int buf = s & 1;
    if (wv == 0) {
      if (s + 1 < nsteps) {              // prefetch next k-step, then drain
        issue(s + 1, buf ^ 1);           // oldest 2 = current buffer
        __builtin_amdgcn_s_wait_tensorcnt(2);
      } else {
        __builtin_amdgcn_s_wait_tensorcnt(0);
      }
    }
    __syncthreads();                     // tile `buf` ready for all waves

    // ---- A fragment (contiguous k = 16*g + s, same order as B) ----
    bf16x16 a;
    {
      int m = wv * 16 + (lane & 15);
      int g = lane >> 4;
#pragma unroll
      for (int v = 0; v < 8; ++v) {
        int k = 2 * v + 16 * g;
        a[2 * v]     = As[buf][m][k];
        a[2 * v + 1] = As[buf][m][k + 1];
      }
    }
#pragma unroll
    for (int j = 0; j < 4; ++j) {
      bf16x16 b;
      int n = j * 16 + (lane & 15);
      int g = lane >> 4;
#pragma unroll
      for (int v = 0; v < 8; ++v) {
        int kk = g * 16 + 2 * v;
        b[2 * v]     = Bs[buf][n][kk];
        b[2 * v + 1] = Bs[buf][n][kk + 1];
      }
      acc[j] = wmma_bf16(a, b, acc[j]);
    }
    __syncthreads();                     // all reads of `buf` done before reuse
  }

  // ---- epilogue: +bias, fp32 store. C-tile: M = v + 8*(lane>=16), N = lane%16
#pragma unroll
  for (int j = 0; j < 4; ++j) {
    int n = c0 + j * 16 + (lane & 15);
    float bv = bias[n];
#pragma unroll
    for (int v = 0; v < 8; ++v) {
      long row = r0 + wv * 16 + v + 8 * (lane >> 4);
      C[row * N + n] = acc[j][v] + bv;
    }
  }
}

// ---------------------------------------------------------------------------
// Kernel: LSTM recurrence. One workgroup (4 waves) owns 16 sequences for all
// T steps; h (bf16) and c (f32) live in LDS; z = xw_t + h@U via WMMA.
// ---------------------------------------------------------------------------
template <int H>
__global__ void __launch_bounds__(128)
k_lstm(const float* __restrict__ xw,     // [NSEQP*T][4H] (bias pre-added)
       const __bf16* __restrict__ UT,    // [4H][H] bf16 transposed
       __bf16* __restrict__ seq_out,     // [NSEQP*T][H] bf16 or nullptr
       float* __restrict__ last_out) {   // [NSEQP][H] fp32 or nullptr
  constexpr int G  = 4 * H;
  constexpr int KS = H / 32;
  constexpr int NT = H / 16;

  __shared__ __bf16 hbf[16][H];
  __shared__ float  cbuf[16][H];
  __shared__ float  zbuf[16][G];

  const int tid  = threadIdx.x;
  const int lane = tid & 31;
  const int wv   = tid >> 5;
  const int seq0 = blockIdx.x * 16;
  const int mrow = lane & 15;
  const int g    = lane >> 4;

  for (int idx = tid; idx < 16 * H; idx += 128) {
    ((__bf16*)hbf)[idx] = (__bf16)0.0f;
    ((float*)cbuf)[idx] = 0.0f;
  }
  __syncthreads();

  const unsigned int* UT32 = (const unsigned int*)UT;

  for (int t = 0; t < TT; ++t) {
    bf16x16 afrag[KS];
#pragma unroll
    for (int ks = 0; ks < KS; ++ks) {
#pragma unroll
      for (int v = 0; v < 8; ++v) {
        int k = ks * 32 + 2 * v + 16 * g;
        afrag[ks][2 * v]     = hbf[mrow][k];
        afrag[ks][2 * v + 1] = hbf[mrow][k + 1];
      }
    }
#pragma unroll
    for (int j = 0; j < NT; ++j) {
      int n = wv * (G / 4) + j * 16 + mrow;
      f32x8 acc;
#pragma unroll
      for (int v = 0; v < 8; ++v) {
        long row = (long)(seq0 + v + 8 * g) * TT + t;
        acc[v] = xw[row * G + n];
      }
#pragma unroll
      for (int ks = 0; ks < KS; ++ks) {
        bf16x16 b;
#pragma unroll
        for (int v = 0; v < 8; ++v) {
          unsigned int w = UT32[((long)n * H + ks * 32 + g * 16 + 2 * v) >> 1];
          union { unsigned int u; __bf16 h[2]; } cv; cv.u = w;
          b[2 * v]     = cv.h[0];
          b[2 * v + 1] = cv.h[1];
        }
        acc = wmma_bf16(afrag[ks], b, acc);
      }
#pragma unroll
      for (int v = 0; v < 8; ++v) zbuf[v + 8 * g][n] = acc[v];
    }
    __syncthreads();

    for (int idx = tid; idx < 16 * H; idx += 128) {
      int m = idx / H, jj = idx % H;
      float zi = zbuf[m][jj];
      float zf = zbuf[m][jj + H];
      float zg = zbuf[m][jj + 2 * H];
      float zo = zbuf[m][jj + 3 * H];
      float cp = cbuf[m][jj];
      float cn = sigmoidf(zf) * cp + sigmoidf(zi) * tanhf(zg);
      float hn = sigmoidf(zo) * tanhf(cn);
      cbuf[m][jj] = cn;
      hbf[m][jj]  = (__bf16)hn;
      if (seq_out)
        seq_out[((long)(seq0 + m) * TT + t) * H + jj] = (__bf16)hn;
      if (last_out && t == TT - 1)
        last_out[(long)(seq0 + m) * H + jj] = hn;
    }
    __syncthreads();
  }
}

// ---------------------------------------------------------------------------
// Kernel: dense(32)+ReLU+BatchNorm heads. enc rows: s[0,200) q[200,400)
// u[400,912) m[912,920). Output rows 0..911 == (d_s | d_q | d_u).
// ---------------------------------------------------------------------------
__global__ void k_head(const float* __restrict__ enc,
    const float* W1, const float* b1, const float* g1, const float* be1,
    const float* m1, const float* v1,
    const float* W2, const float* b2, const float* g2, const float* be2,
    const float* m2, const float* v2,
    const float* W3, const float* b3, const float* g3, const float* be3,
    const float* m3, const float* v3,
    float* __restrict__ out) {
  int idx = blockIdx.x * blockDim.x + threadIdx.x;
  if (idx >= 912 * 32) return;
  int r = idx >> 5, col = idx & 31;
  const float *W, *b, *gg, *be, *mm, *vv;
  int bidx;
  if (r < 200)      { W = W1; b = b1; gg = g1; be = be1; mm = m1; vv = v1;
                      bidx = r / 25; }
  else if (r < 400) { W = W2; b = b2; gg = g2; be = be2; mm = m2; vv = v2;
                      bidx = (r - 200) / 25; }
  else              { W = W3; b = b3; gg = g3; be = be3; mm = m3; vv = v3;
                      bidx = (r - 400) / 64; }
  const float* x1 = enc + (long)r * H2;
  const float* x2 = enc + (long)(912 + bidx) * H2;
  float s = 0.0f;
#pragma unroll 4
  for (int k = 0; k < H2; ++k) s += x1[k] * W[k * 32 + col];
#pragma unroll 4
  for (int k = 0; k < H2; ++k) s += x2[k] * W[(H2 + k) * 32 + col];
  float y = fmaxf(s + b[col], 0.0f);
  out[idx] = gg[col] * (y - mm[col]) * rsqrtf(vv[col] + 1e-3f) + be[col];
}

// ---------------------------------------------------------------------------
// Host: stream-ordered pipeline; scratch in d_ws (hseq aliases dead Xbf).
// ---------------------------------------------------------------------------
extern "C" void kernel_launch(void* const* d_in, const int* in_sizes, int n_in,
                              void* d_out, int out_size, void* d_ws,
                              size_t ws_size, hipStream_t stream) {
  (void)in_sizes; (void)n_in; (void)out_size; (void)ws_size;

  const float* support   = (const float*)d_in[0];
  const float* query     = (const float*)d_in[1];
  const float* unlabel   = (const float*)d_in[2];
  const float* model_ans = (const float*)d_in[3];
  const float* W1 = (const float*)d_in[4];
  const float* U1 = (const float*)d_in[5];
  const float* b1 = (const float*)d_in[6];
  const float* W2 = (const float*)d_in[7];
  const float* U2 = (const float*)d_in[8];
  const float* b2 = (const float*)d_in[9];

  char* ws = (char*)d_ws;
  size_t off = 0;
  auto alloc = [&](size_t bytes) -> void* {
    void* p = ws + off;
    off += (bytes + 255) & ~(size_t)255;
    return p;
  };
  __bf16* W1T = (__bf16*)alloc((size_t)512 * 768 * 2);
  __bf16* U1T = (__bf16*)alloc((size_t)512 * 128 * 2);
  __bf16* W2T = (__bf16*)alloc((size_t)256 * 128 * 2);
  __bf16* U2T = (__bf16*)alloc((size_t)256 * 64 * 2);
  __bf16* Xbf = (__bf16*)alloc((size_t)MPAD * EE * 2);   // layer-1 A (bf16)
  float*  xw  = (float*)alloc((size_t)MPAD * 512 * 4);   // reused for layer 2
  float*  enc = (float*)alloc((size_t)NSEQP * 64 * 4);
  __bf16* hseq = Xbf;  // Xbf dead after GEMM1; reuse for hseq [MPAD][128] bf16

  // 1) weights -> transposed bf16
  k_transpose_bf16<<<(768 * 512 + 255) / 256, 256, 0, stream>>>(W1, W1T, 768, 512);
  k_transpose_bf16<<<(128 * 512 + 255) / 256, 256, 0, stream>>>(U1, U1T, 128, 512);
  k_transpose_bf16<<<(128 * 256 + 255) / 256, 256, 0, stream>>>(W2, W2T, 128, 256);
  k_transpose_bf16<<<( 64 * 256 + 255) / 256, 256, 0, stream>>>(U2, U2T,  64, 256);

  // 2) gather all inputs -> padded bf16 X
  {
    long n = (long)MPAD * EE;
    k_gather_bf16<<<(unsigned)((n + 255) / 256), 256, 0, stream>>>(
        support, query, unlabel, model_ans, Xbf);
  }

  // 3) xw1 = X @ W1 + b1 (TDM-fed WMMA GEMM)
  dim3 grid1(512 / 64, MPAD / 64);
  k_gemm_tdm<<<grid1, 128, 0, stream>>>(Xbf, W1T, b1, xw, EE, 4 * H1);

  // 4) LSTM1 recurrence (return_seq, bf16 out) -> hseq
  k_lstm<H1><<<NSEQP / 16, 128, 0, stream>>>(xw, U1T, hseq, nullptr);

  // 5) xw2 = hseq @ W2 + b2 (same TDM GEMM; xw buffer reused)
  dim3 grid2(256 / 64, MPAD / 64);
  k_gemm_tdm<<<grid2, 128, 0, stream>>>(hseq, W2T, b2, xw, H1, 4 * H2);

  // 6) LSTM2 recurrence (last state only) -> enc
  k_lstm<H2><<<NSEQP / 16, 128, 0, stream>>>(xw, U2T, (__bf16*)nullptr, enc);

  // 7) heads -> d_out (d_s | d_q | d_u)
  k_head<<<(912 * 32 + 127) / 128, 128, 0, stream>>>(
      enc,
      (const float*)d_in[10], (const float*)d_in[11], (const float*)d_in[12],
      (const float*)d_in[13], (const float*)d_in[14], (const float*)d_in[15],
      (const float*)d_in[16], (const float*)d_in[17], (const float*)d_in[18],
      (const float*)d_in[19], (const float*)d_in[20], (const float*)d_in[21],
      (const float*)d_in[22], (const float*)d_in[23], (const float*)d_in[24],
      (const float*)d_in[25], (const float*)d_in[26], (const float*)d_in[27],
      (float*)d_out);
}
